// MultiheadSelfAttention_44478681317699
// MI455X (gfx1250) — matrix operand order repaired and checked by reading, not compiled, not run on hardware
//
#include <hip/hip_runtime.h>

// ---------------------------------------------------------------------------
// MHA forward for MI455X (gfx1250): bf16 WMMA, double-buffered async-to-LDS
// GEMM staging (copy overlapped with WMMA), fused RoPE, transposed-score
// flash attention (in-lane softmax reductions), V and O kept transposed so
// every fragment is a contiguous 16/32-byte load.
// ---------------------------------------------------------------------------

typedef __attribute__((ext_vector_type(16))) __bf16        v16bf;
typedef __attribute__((ext_vector_type(8)))  float         v8f;
typedef __attribute__((ext_vector_type(8)))  unsigned int  v8u;
typedef __attribute__((ext_vector_type(4)))  unsigned int  v4u;
typedef int v4i __attribute__((vector_size(16)));   // matches async builtin param

#define D_MODEL 1024
#define SEQ     2048
#define BATCH   2
#define NHEAD   16
#define DK      64
#define KSTEPS  (D_MODEL / 32)

#define AS1 __attribute__((address_space(1)))
#define AS3 __attribute__((address_space(3)))

#if defined(__AMDGCN__) && __has_builtin(__builtin_amdgcn_global_load_async_to_lds_b128)
#define HAVE_ASYNC_LDS 1
#else
#define HAVE_ASYNC_LDS 0
#endif

__device__ __forceinline__ unsigned short f2bf(float x) {
  unsigned int u = __float_as_uint(x);
  u += 0x7FFFu + ((u >> 16) & 1u);           // round-to-nearest-even
  return (unsigned short)(u >> 16);
}
__device__ __forceinline__ unsigned int pack2(float a, float b) {
  return (unsigned int)f2bf(a) | ((unsigned int)f2bf(b) << 16);
}

union Frag  { v16bf bf; v4u u[2]; };
union Frag1 { v16bf bf; v8u u; };

// B-fragment from global (32B aligned): one 32-byte load -> adjacent VGPRs.
__device__ __forceinline__ v16bf ldBg(const unsigned short* p) {
  Frag1 f;
  f.u = *(const v8u*)(p);
  return f.bf;
}
// B-fragment from LDS (only 16B aligned rows): two b128 loads.
__device__ __forceinline__ v16bf ldB(const unsigned short* p) {
  Frag f;
  f.u[0] = *(const v4u*)(p);
  f.u[1] = *(const v4u*)(p + 8);
  return f.bf;
}
// A-fragment: lane = row (lane&15); chunks k = {16*hi..+7, 16*hi+8..+15}.
__device__ __forceinline__ v16bf ldA(const unsigned short* row, int hi) {
  Frag f;
  f.u[0] = *(const v4u*)(row + 16 * hi);
  f.u[1] = *(const v4u*)(row + 16 * hi + 8);
  return f.bf;
}
__device__ __forceinline__ v8f wmma_bf16(v16bf a, v16bf b, v8f c) {
  return __builtin_amdgcn_wmma_f32_16x16x32_bf16(false, a, false, b,
                                                 (short)0, c, false, false);
}
template <int N> __device__ __forceinline__ void wait_async_le() {
#if HAVE_ASYNC_LDS
#if __has_builtin(__builtin_amdgcn_s_wait_asynccnt)
  __builtin_amdgcn_s_wait_asynccnt(N);
#else
  asm volatile("s_wait_asynccnt %0" ::"i"(N) : "memory");
#endif
#endif
}

// ---------------------------------------------------------------------------
// f32 -> bf16 conversion, 8 elements / thread
// ---------------------------------------------------------------------------
__global__ void __launch_bounds__(256) cvt_bf16(const float* __restrict__ in,
                                                unsigned short* __restrict__ out,
                                                int n) {
  int i = (blockIdx.x * blockDim.x + threadIdx.x) * 8;
  if (i >= n) return;
  const float4* p = (const float4*)(in + i);
  float4 f0 = p[0], f1 = p[1];
  v4u o;
  o[0] = pack2(f0.x, f0.y);
  o[1] = pack2(f0.z, f0.w);
  o[2] = pack2(f1.x, f1.y);
  o[3] = pack2(f1.z, f1.w);
  *(v4u*)(out + i) = o;
}

// ---------------------------------------------------------------------------
// GEMM: out[r][n] = sum_k A[r][k] * W[n][k]   (A @ W^T)
// Block = 4 waves; block tile 128(M) x 64(N); wave tile 32x64.
// A (128x32) + B (64x32) double-buffered in LDS; async copy of tile i+1
// overlaps the WMMAs of tile i (s_wait_asynccnt 6 pipeline).
// MODE 0: Q (RoPE + 1/8, bf16 [B,H,S,64])   MODE 1: K (RoPE, bf16 [B,H,S,64])
// MODE 2: V (bf16 transposed [B,H,64,S])    MODE 3: out-proj (f32 [B,S,1024])
// ---------------------------------------------------------------------------
template <int MODE>
__global__ void __launch_bounds__(128) gemm16(const unsigned short* __restrict__ A,
                                              const unsigned short* __restrict__ W,
                                              unsigned short* __restrict__ outb,
                                              float* __restrict__ outf) {
  __shared__ unsigned short smA[2][128 * 40];   // 80B row stride, 16B chunks
  __shared__ unsigned short smB[2][64 * 40];
  const int tid  = threadIdx.x;
  const int lane = tid & 31;
  const int wv   = tid >> 5;
  const int hi   = lane >> 4;
  const int ln15 = lane & 15;
  const int M0   = blockIdx.x * 128;
  const int N0   = blockIdx.y * 64;

  v8f acc[2][4];
#pragma unroll
  for (int i = 0; i < 2; ++i)
#pragma unroll
    for (int j = 0; j < 4; ++j) acc[i][j] = (v8f)0.0f;

  const int rB = tid >> 1;                    // staged B row 0..63
  const int cB = (tid & 1) * 16;              // 0 / 16

  auto srcA_at = [&](int k0) -> const unsigned short* {
    if constexpr (MODE == 3) {                // gather from [B,H,S,64]
      int h = k0 >> 6, d = k0 & 63;
      int r = M0 + tid;
      int b = r >> 11, s = r & (SEQ - 1);
      return A + (((size_t)(b * NHEAD + h) * SEQ + s) * DK + d);
    } else {
      return A + ((size_t)(M0 + tid) * D_MODEL + k0);
    }
  };
  auto srcB_at = [&](int k0) -> const unsigned short* {
    return W + ((size_t)(N0 + rB) * D_MODEL + k0 + cB);
  };

#if HAVE_ASYNC_LDS
  auto stage_async = [&](int buf, int k0) {
    const unsigned short* sA = srcA_at(k0);
    const unsigned short* sB = srcB_at(k0);
    unsigned short* dA = &smA[buf][tid * 40];
    unsigned short* dB = &smB[buf][rB * 40 + cB];
#pragma unroll
    for (int q = 0; q < 4; ++q)
      __builtin_amdgcn_global_load_async_to_lds_b128(
          (AS1 v4i*)(sA + 8 * q), (AS3 v4i*)(dA + 8 * q), 0, 0);
    __builtin_amdgcn_global_load_async_to_lds_b128(
        (AS1 v4i*)(sB), (AS3 v4i*)(dB), 0, 0);
    __builtin_amdgcn_global_load_async_to_lds_b128(
        (AS1 v4i*)(sB + 8), (AS3 v4i*)(dB + 8), 0, 0);
  };
  stage_async(0, 0);                          // prologue: tile 0 in flight
#endif

  for (int it = 0; it < KSTEPS; ++it) {
    const int buf = it & 1;
#if HAVE_ASYNC_LDS
    if (it + 1 < KSTEPS) {
      stage_async(buf ^ 1, (it + 1) * 32);    // next tile under this tile's math
      wait_async_le<6>();                     // tile 'it' (older 6) has landed
    } else {
      wait_async_le<0>();
    }
    if (it + 2 < KSTEPS) {                    // warm L2 two tiles ahead
      __builtin_prefetch(srcA_at((it + 2) * 32), 0, 1);
      __builtin_prefetch(srcB_at((it + 2) * 32), 0, 1);
    }
    __syncthreads();                          // everyone's tile-'it' data ready
#else
    const unsigned short* sA = srcA_at(it * 32);
    const unsigned short* sB = srcB_at(it * 32);
    v4u a0 = *(const v4u*)(sA);
    v4u a1 = *(const v4u*)(sA + 8);
    v4u a2 = *(const v4u*)(sA + 16);
    v4u a3 = *(const v4u*)(sA + 24);
    v4u b0 = *(const v4u*)(sB);
    v4u b1 = *(const v4u*)(sB + 8);
    __syncthreads();
    *(v4u*)&smA[0][tid * 40 + 0]  = a0;
    *(v4u*)&smA[0][tid * 40 + 8]  = a1;
    *(v4u*)&smA[0][tid * 40 + 16] = a2;
    *(v4u*)&smA[0][tid * 40 + 24] = a3;
    *(v4u*)&smB[0][rB * 40 + cB]     = b0;
    *(v4u*)&smB[0][rB * 40 + cB + 8] = b1;
    __syncthreads();
#endif
    const int cbuf = HAVE_ASYNC_LDS ? buf : 0;
    v16bf af[2];
#pragma unroll
    for (int i = 0; i < 2; ++i)
      af[i] = ldA(&smA[cbuf][(wv * 32 + i * 16 + ln15) * 40], hi);
#pragma unroll
    for (int j = 0; j < 4; ++j) {
      v16bf bf = ldB(&smB[cbuf][(j * 16 + ln15) * 40 + 16 * hi]);
#pragma unroll
      for (int i = 0; i < 2; ++i) acc[i][j] = wmma_bf16(af[i], bf, acc[i][j]);
    }
    __syncthreads();                          // reads done before buffer reuse
  }

  const int h = N0 >> 6;                      // head index (MODE 0..2)
#pragma unroll
  for (int j = 0; j < 4; ++j) {
    int nloc = j * 16 + ln15;                 // 0..63 column within tile
    // RoPE constants depend only on the column -> hoisted out of i loop
    float inv  = __builtin_exp2f(-(float)(nloc >> 1) * 0.41524101f); // log2(1e4)/32
    bool  even = (nloc & 1) == 0;
#pragma unroll
    for (int i = 0; i < 2; ++i) {
      if constexpr (MODE == 0 || MODE == 1) {
#pragma unroll
        for (int v = 0; v < 8; ++v) {
          int r = M0 + wv * 32 + i * 16 + v + 8 * hi;
          int b = r >> 11, s = r & (SEQ - 1);
          float sn, cs;
          __sincosf((float)s * inv, &sn, &cs);
          float mine  = acc[i][j][v];
          float other = __shfl_xor(mine, 1, 32);
          float val = even ? (mine * cs - other * sn)
                           : (mine * cs + other * sn);
          if constexpr (MODE == 0) val *= 0.125f;   // 1/sqrt(64)
          outb[(((size_t)(b * NHEAD + h) * SEQ + s) * DK) + nloc] = f2bf(val);
        }
      } else if constexpr (MODE == 2) {
        v4u pk;
#pragma unroll
        for (int v = 0; v < 8; v += 2)
          pk[v >> 1] = pack2(acc[i][j][v], acc[i][j][v + 1]);
        int r0 = M0 + wv * 32 + i * 16 + 8 * hi;
        int b = r0 >> 11, s0 = r0 & (SEQ - 1);
        *(v4u*)(outb + ((size_t)(b * NHEAD + h) * DK + nloc) * SEQ + s0) = pk;
      } else {
#pragma unroll
        for (int v = 0; v < 8; ++v) {
          int r = M0 + wv * 32 + i * 16 + v + 8 * hi;
          outf[(size_t)r * D_MODEL + N0 + nloc] = acc[i][j][v];
        }
      }
    }
  }
}

// ---------------------------------------------------------------------------
// Flash attention, fully transposed: S^T = K Q^T, O^T = V^T P^T.
// A lane owns one query column -> softmax max/sum are in-lane reductions
// plus a single xor-16 half-merge each. rm/rs/corr are per-lane scalars.
// ---------------------------------------------------------------------------
__global__ void __launch_bounds__(128) flash_attn(const unsigned short* __restrict__ Q,
                                                  const unsigned short* __restrict__ K,
                                                  const unsigned short* __restrict__ Vt,
                                                  unsigned short* __restrict__ O) {
  __shared__ unsigned short smP[4][16 * 40];  // per-wave P^T [query][key]
  const int tid  = threadIdx.x;
  const int lane = tid & 31;
  const int wv   = tid >> 5;
  const int hi   = lane >> 4;
  const int ln15 = lane & 15;
  const int bh   = blockIdx.y;                      // b*16 + h
  const int q0   = blockIdx.x * 64 + wv * 16;       // wave's first query row

  // Q^T as B-fragments (lane = query column), loaded once
  v16bf bq[2];
  {
    const unsigned short* qrow = Q + ((size_t)bh * SEQ + q0 + ln15) * DK;
    bq[0] = ldBg(qrow + 16 * hi);           // d = 0..31
    bq[1] = ldBg(qrow + 32 + 16 * hi);      // d = 32..63
  }

  v8f ot[4];                                // O^T: frag j = d rows j*16..+15
#pragma unroll
  for (int j = 0; j < 4; ++j) ot[j] = (v8f)0.0f;
  float rm = -1e30f, rs = 0.0f;             // per-lane (per-query) stats
  const int qpos = q0 + ln15;

  unsigned int* smPu = (unsigned int*)&smP[wv][0];

  const int qend = q0 + 15;
  for (int kt = 0; kt <= qend; kt += 32) {
    // ---- S^T tiles: rows = keys, cols = queries -------------------------
    v8f st[2];
#pragma unroll
    for (int t = 0; t < 2; ++t) {
      const unsigned short* krow =
          K + ((size_t)bh * SEQ + kt + t * 16 + ln15) * DK;
      v16bf ak0 = ldA(krow, hi);            // d = 0..31
      v16bf ak1 = ldA(krow + 32, hi);       // d = 32..63
      v8f a = (v8f)0.0f;
      a = wmma_bf16(ak0, bq[0], a);
      a = wmma_bf16(ak1, bq[1], a);
      st[t] = a;
    }

    // ---- causal mask + in-lane max over this lane's 16 keys -------------
    float mx = -1e30f;
#pragma unroll
    for (int t = 0; t < 2; ++t)
#pragma unroll
      for (int v = 0; v < 8; ++v) {
        int kpos = kt + t * 16 + v + 8 * hi;
        float s = (kpos <= qpos) ? st[t][v] : -1.0e6f;
        st[t][v] = s;
        mx = fmaxf(mx, s);
      }
    mx = fmaxf(mx, __shfl_xor(mx, 16, 32)); // merge the two key halves
    float nm   = fmaxf(rm, mx);
    float corr = __expf(rm - nm);

    // ---- exp + in-lane sum ----------------------------------------------
    float ps = 0.0f;
#pragma unroll
    for (int t = 0; t < 2; ++t)
#pragma unroll
      for (int v = 0; v < 8; ++v) {
        float p = __expf(st[t][v] - nm);
        st[t][v] = p;
        ps += p;
      }
    ps += __shfl_xor(ps, 16, 32);
    rs = rs * corr + ps;
    rm = nm;
#pragma unroll
    for (int j = 0; j < 4; ++j) ot[j] *= corr;   // per-lane scalar rescale

    // ---- P^T -> LDS [query][key], keys consecutive per lane: pack pairs -
#pragma unroll
    for (int t = 0; t < 2; ++t)
#pragma unroll
      for (int v = 0; v < 8; v += 2)
        smPu[ln15 * 20 + t * 8 + 4 * hi + (v >> 1)] =
            pack2(st[t][v], st[t][v + 1]);

    // B-fragment of P^T: lane = query col, 16 contiguous keys at 16*hi
    v16bf bp = ldB(&smP[wv][ln15 * 40 + 16 * hi]);

    // ---- O^T += V^T P^T : A = V^T rows (d), contiguous keys -------------
#pragma unroll
    for (int j = 0; j < 4; ++j) {
      const unsigned short* vrow =
          Vt + ((size_t)bh * DK + j * 16 + ln15) * SEQ + kt;
      v16bf av = ldA(vrow, hi);
      ot[j] = wmma_bf16(av, bp, ot[j]);
    }
  }

  // ---- normalize, 8 consecutive d per lane -> one b128 store per frag ---
  float inv = 1.0f / rs;
  int s = qpos;
#pragma unroll
  for (int j = 0; j < 4; ++j) {
    v4u pk;
#pragma unroll
    for (int v = 0; v < 8; v += 2)
      pk[v >> 1] = pack2(ot[j][v] * inv, ot[j][v + 1] * inv);
    *(v4u*)(O + ((size_t)bh * SEQ + s) * DK + j * 16 + 8 * hi) = pk;
  }
}

// ---------------------------------------------------------------------------
// Launcher
// ---------------------------------------------------------------------------
extern "C" void kernel_launch(void* const* d_in, const int* in_sizes, int n_in,
                              void* d_out, int out_size, void* d_ws, size_t ws_size,
                              hipStream_t stream) {
  (void)in_sizes; (void)n_in; (void)out_size; (void)ws_size;
  const float* x  = (const float*)d_in[0];
  const float* wq = (const float*)d_in[1];
  const float* wk = (const float*)d_in[2];
  const float* wv = (const float*)d_in[3];
  const float* wo = (const float*)d_in[4];
  float* out = (float*)d_out;

  char* ws = (char*)d_ws;                     // 48 MB used, L2-resident
  unsigned short* xb  = (unsigned short*)(ws);
  unsigned short* wqb = (unsigned short*)(ws + ((size_t)8  << 20));
  unsigned short* wkb = (unsigned short*)(ws + ((size_t)10 << 20));
  unsigned short* wvb = (unsigned short*)(ws + ((size_t)12 << 20));
  unsigned short* wob = (unsigned short*)(ws + ((size_t)14 << 20));
  unsigned short* Qb  = (unsigned short*)(ws + ((size_t)16 << 20));
  unsigned short* Kb  = (unsigned short*)(ws + ((size_t)24 << 20));
  unsigned short* Vtb = (unsigned short*)(ws + ((size_t)32 << 20));
  unsigned short* Ob  = (unsigned short*)(ws + ((size_t)40 << 20));

  const int NX = BATCH * SEQ * D_MODEL;       // 4,194,304
  const int NW = D_MODEL * D_MODEL;           // 1,048,576
  cvt_bf16<<<NX / 2048, 256, 0, stream>>>(x,  xb,  NX);
  cvt_bf16<<<NW / 2048, 256, 0, stream>>>(wq, wqb, NW);
  cvt_bf16<<<NW / 2048, 256, 0, stream>>>(wk, wkb, NW);
  cvt_bf16<<<NW / 2048, 256, 0, stream>>>(wv, wvb, NW);
  cvt_bf16<<<NW / 2048, 256, 0, stream>>>(wo, wob, NW);

  dim3 g(32, 16), b(128);
  gemm16<0><<<g, b, 0, stream>>>(xb, wqb, Qb,  nullptr);   // Q + RoPE + scale
  gemm16<1><<<g, b, 0, stream>>>(xb, wkb, Kb,  nullptr);   // K + RoPE
  gemm16<2><<<g, b, 0, stream>>>(xb, wvb, Vtb, nullptr);   // V (transposed)
  flash_attn<<<dim3(32, 32), b, 0, stream>>>(Qb, Kb, Vtb, Ob);
  gemm16<3><<<g, b, 0, stream>>>(Ob, wob, nullptr, out);   // output proj
}
